// MixerAttention_8770323219185
// MI455X (gfx1250) — matrix operand order repaired and checked by reading, not compiled, not run on hardware
//
#include <hip/hip_runtime.h>
#include <hip/hip_bf16.h>

// ---------------------------------------------------------------------------
// MixerAttention for MI455X (gfx1250, wave32, WMMA).
// Matrix math: v_wmma_f32_16x16x32_bf16 (bf16 in, f32 acc).
// Tile movement: global_load_async_to_lds_b128 + s_wait_asynccnt (ASYNCcnt),
// double-buffered LDS so WMMA overlaps the next tile's DMA.
// ---------------------------------------------------------------------------

constexpr int Bc   = 2;
constexpr int Tc   = 2048;
constexpr int Dc   = 1024;
constexpr int Hc   = 16;
constexpr int HKVc = 4;
constexpr int DHc  = 64;
constexpr int REPc = Hc / HKVc;                 // 4
constexpr int NQKV = DHc * (Hc + 2 * HKVc);     // 1536
constexpr int NTOT = NQKV + Dc;                 // 2560 (qkv || g fused GEMM)
constexpr int Mc   = Bc * Tc;                   // 4096

typedef __attribute__((ext_vector_type(16))) __bf16 v16bf;
typedef __attribute__((ext_vector_type(8)))  __bf16 v8bf;
typedef __attribute__((ext_vector_type(8)))  float  v8f;

__device__ __forceinline__ v16bf cat8(v8bf lo, v8bf hi) {
  return __builtin_shufflevector(lo, hi, 0,1,2,3,4,5,6,7,8,9,10,11,12,13,14,15);
}

__device__ __forceinline__ v8f wmma_bf16(v16bf a, v16bf b, v8f c) {
  // (neg_a, A, neg_b, B, c_mod, C, reuse_a, reuse_b)
  return __builtin_amdgcn_wmma_f32_16x16x32_bf16(false, a, false, b, (short)0, c,
                                                 false, false);
}

// Async 16B copy global -> LDS (GLOBAL_LOAD_ASYNC_TO_LDS_B128, ASYNCcnt).
// LDS operand = low 32 bits of the flat shared address == LDS byte offset
__device__ __forceinline__ void async_ld_b128(const __bf16* lds_dst,
                                              const __bf16* g_src) {
  unsigned off = (unsigned)(unsigned long long)lds_dst;
  asm volatile("global_load_async_to_lds_b128 %0, %1, off"
               :: "v"(off), "v"(g_src)
               : "memory");
}
#define ASYNC_WAIT_ALL()  asm volatile("s_wait_asynccnt 0x0" ::: "memory")
#define ASYNC_WAIT_TILE() asm volatile("s_wait_asynccnt 0x4" ::: "memory")

// ---------------------------------------------------------------------------
// Kernel 1: fp32 -> bf16 conversion. x row-major [M][K]; weights transposed +
// concatenated as Wt[n][k] (k contiguous) so GEMM B-fragments (fixed N,
// K 8h..8h+7 & 16+8h..+7 in lane half h; ISA 7.12.2) load straight from LDS.
// ---------------------------------------------------------------------------
__global__ __launch_bounds__(256)
void prep_kernel(const float* __restrict__ x, const float* __restrict__ w_qkv,
                 const float* __restrict__ w_g,
                 __bf16* __restrict__ xb, __bf16* __restrict__ wt)
{
  size_t tid0   = (size_t)blockIdx.x * blockDim.x + threadIdx.x;
  size_t stride = (size_t)gridDim.x * blockDim.x;

  for (size_t i = tid0; i < (size_t)Mc * Dc; i += stride)
    xb[i] = (__bf16)x[i];

  for (size_t i = tid0; i < (size_t)Dc * NQKV; i += stride) {
    size_t kk = i / NQKV, n = i % NQKV;            // coalesced read of w_qkv
    wt[n * Dc + kk] = (__bf16)w_qkv[i];
  }
  for (size_t i = tid0; i < (size_t)Dc * Dc; i += stride) {
    size_t kk = i / Dc, n = i % Dc;
    wt[(NQKV + n) * Dc + kk] = (__bf16)w_g[i];
  }
}

// ---------------------------------------------------------------------------
// Kernel 2: C[M][N] = Xbf16[M][K] * Wt[N][K]^T, f32 accumulation.
// 128x128 block tile, K-step 32, 8 waves (4x2), 32x64 sub-tile per wave ->
// 8 WMMA / K-step. Double-buffered LDS fed by async-to-LDS copies; the next
// K-tile streams in while the current one is consumed by WMMA.
// ---------------------------------------------------------------------------
#define GMT 128
#define GNT 128
#define GKT 32

__global__ __launch_bounds__(256)
void gemm_bf16_kernel(const __bf16* __restrict__ A,   // [M][K]
                      const __bf16* __restrict__ Bt,  // [N][K]
                      float* __restrict__ C,          // [M][N]
                      int Mdim, int Ndim, int Kdim)
{
  __shared__ __align__(16) __bf16 As[2][GMT][GKT];
  __shared__ __align__(16) __bf16 Bs[2][GNT][GKT];

  const int tid  = threadIdx.x;
  const int lane = tid & 31, wave = tid >> 5;
  const int wm = wave >> 1, wn = wave & 1;     // 4x2 wave grid
  const int l16 = lane & 15, half = lane >> 4;
  const int m0 = blockIdx.x * GMT;
  const int n0 = blockIdx.y * GNT;

  v8f acc[2][4] = {};

  const int ldr = tid >> 1;             // 0..127 (tile row)
  const int ldc = (tid & 1) * 16;       // 0 / 16
  const __bf16* gaBase = A  + (size_t)(m0 + ldr) * Kdim + ldc;
  const __bf16* gbBase = Bt + (size_t)(n0 + ldr) * Kdim + ldc;

  auto issue_tile = [&](int k0, int buf) {
    async_ld_b128(&As[buf][ldr][ldc],     gaBase + k0);
    async_ld_b128(&As[buf][ldr][ldc + 8], gaBase + k0 + 8);
    async_ld_b128(&Bs[buf][ldr][ldc],     gbBase + k0);
    async_ld_b128(&Bs[buf][ldr][ldc + 8], gbBase + k0 + 8);
  };

  const int nk = Kdim / GKT;
  issue_tile(0, 0);

  for (int i = 0; i < nk; ++i) {
    const int buf = i & 1;
    if (i + 1 < nk) {
      issue_tile((i + 1) * GKT, (i + 1) & 1);
      ASYNC_WAIT_TILE();                 // current tile done; next in flight
    } else {
      ASYNC_WAIT_ALL();
    }
    __syncthreads();

    // A fragments (16x32 bf16): half0 holds K 0..7 & 16..23, half1 the rest.
    v16bf af[2];
    #pragma unroll
    for (int rt = 0; rt < 2; ++rt) {
      const int r = wm * 32 + rt * 16 + l16;
      v8bf lo = *(const v8bf*)&As[buf][r][half * 8];
      v8bf hi = *(const v8bf*)&As[buf][r][16 + half * 8];
      af[rt] = cat8(lo, hi);
    }
    // B fragments (32x16): lane holds column n, K half*8..+8 & 16+half*8..+8.
    v16bf bfm[4];
    #pragma unroll
    for (int ct = 0; ct < 4; ++ct) {
      const int n = wn * 64 + ct * 16 + l16;
      v8bf lo = *(const v8bf*)&Bs[buf][n][half * 8];
      v8bf hi = *(const v8bf*)&Bs[buf][n][16 + half * 8];
      bfm[ct] = cat8(lo, hi);
    }
    #pragma unroll
    for (int rt = 0; rt < 2; ++rt)
      #pragma unroll
      for (int ct = 0; ct < 4; ++ct)
        acc[rt][ct] = wmma_bf16(af[rt], bfm[ct], acc[rt][ct]);

    __syncthreads();                     // protect buf before it is re-filled
  }

  // C layout: VGPR v -> row v + 8*half, column l16.
  #pragma unroll
  for (int rt = 0; rt < 2; ++rt)
    #pragma unroll
    for (int ct = 0; ct < 4; ++ct)
      #pragma unroll
      for (int v = 0; v < 8; ++v) {
        const size_t r = (size_t)(m0 + wm * 32 + rt * 16 + v + 8 * half);
        C[r * Ndim + n0 + wn * 64 + ct * 16 + l16] = acc[rt][ct][v];
      }
}

// ---------------------------------------------------------------------------
// Kernel 3: RMSNorm + RoPE + softmax_scaler*log(pos) (1/sqrt(DH) folded into
// Q). One wave per head-slot; lane l owns RoPE pair (l, l+32).
// V is written TRANSPOSED ([B][HKV][DH][T], t contiguous) so the attention
// kernel's P*V B-fragments stream into LDS with async b128 (no transpose).
// ---------------------------------------------------------------------------
__global__ __launch_bounds__(256)
void qkv_post_kernel(const float* __restrict__ proj,
                     const float* __restrict__ softmax_scaler,
                     __bf16* __restrict__ qb,   // [B][H][T][DH]
                     __bf16* __restrict__ kb,   // [B][HKV][T][DH]
                     __bf16* __restrict__ vb)   // [B][HKV][DH][T]  (transposed)
{
  const int bt = blockIdx.x;
  const int b = bt / Tc, t = bt % Tc;
  const int lane = threadIdx.x & 31, wave = threadIdx.x >> 5;
  const float* row = proj + (size_t)bt * NTOT;

  for (int slot = wave; slot < Hc + 2 * HKVc; slot += 8) {
    const bool isq = slot < Hc;
    const bool isv = slot >= Hc + HKVc;
    int kg, off;
    if (isq)      { kg = slot >> 2;          off = kg * 384 + (slot & 3) * 64; }
    else if (isv) { kg = slot - Hc - HKVc;   off = kg * 384 + 320; }
    else          { kg = slot - Hc;          off = kg * 384 + 256; }

    float x1 = row[off + lane];
    float x2 = row[off + 32 + lane];

    if (isv) {
      __bf16* dst = vb + ((size_t)b * HKVc + kg) * DHc * Tc;
      dst[(size_t)lane * Tc + t]        = (__bf16)x1;   // V^T: [dh][t]
      dst[(size_t)(lane + 32) * Tc + t] = (__bf16)x2;
      continue;
    }

    // RMSNorm over DH=64 via wave32 xor-reduction (2 elems/lane).
    float ss = x1 * x1 + x2 * x2;
    #pragma unroll
    for (int m = 16; m >= 1; m >>= 1) ss += __shfl_xor(ss, m, 32);
    const float r = rsqrtf(ss * (1.0f / 64.0f) + 1e-6f);
    x1 *= r; x2 *= r;

    // RoPE: inv_freq = base^(-(2*lane)/64)
    const float invf = __powf(10000.0f, -((float)(2 * lane)) / 64.0f);
    float s, c;
    __sincosf((float)t * invf, &s, &c);
    float y1 =  x1 * c + x2 * s;
    float y2 = -x1 * s + x2 * c;

    if (isq) {
      const float sc = softmax_scaler[slot] * __logf((float)(t + 1)) * 0.125f;
      y1 *= sc; y2 *= sc;
      __bf16* dst = qb + (((size_t)b * Hc + slot) * Tc + t) * DHc;
      dst[lane]      = (__bf16)y1;
      dst[lane + 32] = (__bf16)y2;
    } else {
      __bf16* dst = kb + (((size_t)b * HKVc + kg) * Tc + t) * DHc;
      dst[lane]      = (__bf16)y1;
      dst[lane + 32] = (__bf16)y2;
    }
  }
}

// ---------------------------------------------------------------------------
// Kernel 4: causal flash attention + fused SiLU gate.
// Block = 128 Q rows, 8 waves x 16 rows. K tiles ([kpos][dh]) and
// pre-transposed V tiles ([dh][kpos]) of 64 stream into double-buffered LDS
// via async-to-LDS; online softmax with 16-lane xor reductions; P staged via
// per-wave LDS for the C->A fragment relayout (intra-wave DS is in-order).
// ---------------------------------------------------------------------------
#define QT    128
#define KTILE 64

__global__ __launch_bounds__(256)
void flash_attn_kernel(const __bf16* __restrict__ qb,
                       const __bf16* __restrict__ kb,
                       const __bf16* __restrict__ vb,   // [B][HKV][DH][T]
                       const float*  __restrict__ proj,
                       float* __restrict__ out)
{
  __shared__ __align__(16) __bf16 Ks[2][KTILE][DHc];    // [kpos][dh]
  __shared__ __align__(16) __bf16 Vt[2][DHc][KTILE];    // [dh][kpos]
  __shared__ __align__(16) __bf16 Ps[8][16][KTILE];     // per-wave P staging

  const int bh = blockIdx.x;
  const int b = bh / Hc, h = bh % Hc;
  const int kg = h / REPc;
  const int q0 = blockIdx.y * QT;
  const int tid = threadIdx.x, lane = tid & 31, wave = tid >> 5;
  const int l16 = lane & 15, half = lane >> 4;

  const __bf16* qbase  = qb + ((size_t)b * Hc   + h ) * Tc * DHc;
  const __bf16* kbase  = kb + ((size_t)b * HKVc + kg) * Tc * DHc;
  const __bf16* vtbase = vb + ((size_t)b * HKVc + kg) * DHc * Tc;

  const int qrow0 = q0 + wave * 16;

  // Q fragments (held in registers for the whole block): DH=64 -> 2 K-chunks.
  v16bf qf[2];
  {
    const __bf16* qp = qbase + (size_t)(qrow0 + l16) * DHc;
    #pragma unroll
    for (int kc = 0; kc < 2; ++kc) {
      v8bf lo = *(const v8bf*)(qp + kc * 32 + half * 8);
      v8bf hi = *(const v8bf*)(qp + kc * 32 + 16 + half * 8);
      qf[kc] = cat8(lo, hi);
    }
  }

  v8f O[4] = {};
  float mrow[8], lrow[8];
  #pragma unroll
  for (int v = 0; v < 8; ++v) { mrow[v] = -3.0e38f; lrow[v] = 0.0f; }

  const int r  = tid >> 2;              // tile row (kpos for K, dh for V^T)
  const int cc = (tid & 3) * 16;
  auto issue_kv = [&](int kt0, int buf) {
    const __bf16* kp = kbase  + (size_t)(kt0 + r) * DHc + cc;
    const __bf16* vp = vtbase + (size_t)r * Tc + kt0 + cc;
    async_ld_b128(&Ks[buf][r][cc],     kp);
    async_ld_b128(&Ks[buf][r][cc + 8], kp + 8);
    async_ld_b128(&Vt[buf][r][cc],     vp);
    async_ld_b128(&Vt[buf][r][cc + 8], vp + 8);
  };

  const int nk = (q0 + QT) / KTILE;     // causal bound for this block
  issue_kv(0, 0);

  for (int i = 0; i < nk; ++i) {
    const int buf = i & 1;
    const int kt0 = i * KTILE;
    if (i + 1 < nk) {
      issue_kv((i + 1) * KTILE, (i + 1) & 1);
      ASYNC_WAIT_TILE();
    } else {
      ASYNC_WAIT_ALL();
    }
    __syncthreads();

    if (kt0 <= qrow0 + 15) {            // skip fully-masked tiles (causal)
      // S = Q * K^T : 4 col-tiles x 2 K-chunks = 8 WMMA.
      v8f S[4] = {};
      #pragma unroll
      for (int c = 0; c < 4; ++c)
        #pragma unroll
        for (int kc = 0; kc < 2; ++kc) {
          v8bf lo = *(const v8bf*)&Ks[buf][c * 16 + l16][kc * 32 + half * 8];
          v8bf hi = *(const v8bf*)&Ks[buf][c * 16 + l16][kc * 32 + 16 + half * 8];
          S[c] = wmma_bf16(qf[kc], cat8(lo, hi), S[c]);
        }

      // Causal mask (softmax scale already folded into Q).
      if (kt0 + KTILE - 1 > qrow0) {
        #pragma unroll
        for (int c = 0; c < 4; ++c)
          #pragma unroll
          for (int v = 0; v < 8; ++v) {
            const int qr = qrow0 + v + 8 * half;
            const int kp = kt0 + c * 16 + l16;
            if (kp > qr) S[c][v] = -3.0e38f;
          }
      }

      // Online softmax (row stats replicated across each 16-lane half).
      float alpha[8];
      #pragma unroll
      for (int v = 0; v < 8; ++v) {
        float mx = fmaxf(fmaxf(S[0][v], S[1][v]), fmaxf(S[2][v], S[3][v]));
        #pragma unroll
        for (int msk = 1; msk < 16; msk <<= 1) mx = fmaxf(mx, __shfl_xor(mx, msk, 32));
        const float mnew = fmaxf(mrow[v], mx);
        alpha[v] = __expf(mrow[v] - mnew);
        float ps = 0.0f;
        #pragma unroll
        for (int c = 0; c < 4; ++c) {
          const float p = __expf(S[c][v] - mnew);
          S[c][v] = p;
          ps += p;
        }
        #pragma unroll
        for (int msk = 1; msk < 16; msk <<= 1) ps += __shfl_xor(ps, msk, 32);
        lrow[v] = lrow[v] * alpha[v] + ps;
        mrow[v] = mnew;
      }

      // Stage P (C layout -> LDS row-major) then reload as A fragments.
      #pragma unroll
      for (int c = 0; c < 4; ++c)
        #pragma unroll
        for (int v = 0; v < 8; ++v)
          Ps[wave][v + 8 * half][c * 16 + l16] = (__bf16)S[c][v];
      v16bf pf[2];
      {
        const __bf16* pp = &Ps[wave][l16][0];
        #pragma unroll
        for (int kc = 0; kc < 2; ++kc) {
          v8bf lo = *(const v8bf*)(pp + kc * 32 + half * 8);
          v8bf hi = *(const v8bf*)(pp + kc * 32 + 16 + half * 8);
          pf[kc] = cat8(lo, hi);
        }
      }

      // O = alpha*O + P * V : 4 dh-tiles x 2 K-chunks = 8 WMMA.
      #pragma unroll
      for (int d = 0; d < 4; ++d) {
        #pragma unroll
        for (int v = 0; v < 8; ++v) O[d][v] *= alpha[v];
        #pragma unroll
        for (int kc = 0; kc < 2; ++kc) {
          v8bf lo = *(const v8bf*)&Vt[buf][d * 16 + l16][kc * 32 + half * 8];
          v8bf hi = *(const v8bf*)&Vt[buf][d * 16 + l16][kc * 32 + 16 + half * 8];
          O[d] = wmma_bf16(pf[kc], cat8(lo, hi), O[d]);
        }
      }
    }
    __syncthreads();                    // protect buf before it is re-filled
  }

  // Epilogue: normalize and fuse SiLU(g) gate; output fp32 [B][T][H][DH].
  float inv_l[8];
  #pragma unroll
  for (int v = 0; v < 8; ++v) inv_l[v] = 1.0f / lrow[v];

  #pragma unroll
  for (int d = 0; d < 4; ++d)
    #pragma unroll
    for (int v = 0; v < 8; ++v) {
      const int qr = qrow0 + v + 8 * half;
      const int dh = d * 16 + l16;
      const size_t rr = (size_t)b * Tc + qr;
      const float g  = proj[rr * NTOT + NQKV + h * DHc + dh];
      const float sg = g / (1.0f + __expf(-g));
      out[(rr * Hc + h) * DHc + dh] = O[d][v] * inv_l[v] * sg;
    }
}

// ---------------------------------------------------------------------------
// Launch. Workspace layout (~66 MiB): xb | wt | proj | qb | kb | vb.
// ---------------------------------------------------------------------------
extern "C" void kernel_launch(void* const* d_in, const int* in_sizes, int n_in,
                              void* d_out, int out_size, void* d_ws, size_t ws_size,
                              hipStream_t stream) {
  const float* x      = (const float*)d_in[0];
  const float* w_qkv  = (const float*)d_in[1];
  const float* sscal  = (const float*)d_in[2];
  const float* w_g    = (const float*)d_in[3];
  float* out          = (float*)d_out;

  char* ws = (char*)d_ws;
  size_t off = 0;
  auto take = [&](size_t bytes) {
    char* p = ws + off;
    off = (off + bytes + 255) & ~(size_t)255;
    return p;
  };
  __bf16* xb   = (__bf16*)take((size_t)Mc * Dc * 2);
  __bf16* wt   = (__bf16*)take((size_t)NTOT * Dc * 2);
  float*  proj = (float*) take((size_t)Mc * NTOT * 4);
  __bf16* qb   = (__bf16*)take((size_t)Bc * Hc   * Tc * DHc * 2);
  __bf16* kb   = (__bf16*)take((size_t)Bc * HKVc * Tc * DHc * 2);
  __bf16* vb   = (__bf16*)take((size_t)Bc * HKVc * Tc * DHc * 2);

  prep_kernel<<<2048, 256, 0, stream>>>(x, w_qkv, w_g, xb, wt);

  dim3 ggrid(Mc / GMT, NTOT / GNT);   // 32 x 20
  gemm_bf16_kernel<<<ggrid, 256, 0, stream>>>(xb, wt, proj, Mc, NTOT, Dc);

  qkv_post_kernel<<<Bc * Tc, 256, 0, stream>>>(proj, sscal, qb, kb, vb);

  dim3 agrid(Bc * Hc, Tc / QT);       // 32 x 16
  flash_attn_kernel<<<agrid, 256, 0, stream>>>(qb, kb, vb, proj, out);
}
